// SingleFeatureGNNModel_32152125177973
// MI455X (gfx1250) — matrix operand order repaired and checked by reading, not compiled, not run on hardware
//
#include <hip/hip_runtime.h>
#include <hip/hip_bf16.h>

typedef __attribute__((ext_vector_type(16))) _Float16     v16h;
typedef __attribute__((ext_vector_type(8)))  _Float16     v8h;
typedef __attribute__((ext_vector_type(4)))  _Float16     v4h;
typedef __attribute__((ext_vector_type(8)))  float        v8f;
typedef __attribute__((ext_vector_type(4)))  unsigned int v4u;
typedef __attribute__((ext_vector_type(8)))  int          v8i;
typedef __attribute__((ext_vector_type(4)))  int          v4i;

#define HID   128
#define EPS   1e-5f
#define NPT   16           // nodes per GEMM tile
#define SROW  136          // LDS row stride in halves (128 + 8 pad -> 68 dwords)

__device__ __forceinline__ void atomAdd(float* p, float v) {
    (void)__hip_atomic_fetch_add(p, v, __ATOMIC_RELAXED, __HIP_MEMORY_SCOPE_AGENT);
}

// ---------------------------------------------------------------- degree ----
__global__ void k_init_deg(float* __restrict__ deg, int N) {
    int i = blockIdx.x * blockDim.x + threadIdx.x;
    if (i < N) deg[i] = 1.0f;                       // self-loop
}

__global__ void k_accum_deg(const long long* __restrict__ dst, float* __restrict__ deg, int E) {
    int e = blockIdx.x * blockDim.x + threadIdx.x;
    if (e < E) atomAdd(&deg[dst[e]], 1.0f);
}

__global__ void k_dinv(float* __restrict__ deg, int N) {
    int i = blockIdx.x * blockDim.x + threadIdx.x;
    if (i < N) deg[i] = rsqrtf(deg[i]);             // in place: deg -> dinv
}

// ------------------------------------------------------- one-shot packing ---
// X (f32) -> Xh (f16), 8 elements per thread.
__global__ void k_cvt16(const float* __restrict__ s, v8h* __restrict__ d, int n8) {
    int i = blockIdx.x * blockDim.x + threadIdx.x;
    if (i >= n8) return;
    const float4* sp = (const float4*)s + (size_t)i * 2;
    float4 a = sp[0], b = sp[1];
    v8h o;
    o[0] = (_Float16)a.x; o[1] = (_Float16)a.y; o[2] = (_Float16)a.z; o[3] = (_Float16)a.w;
    o[4] = (_Float16)b.x; o[5] = (_Float16)b.y; o[6] = (_Float16)b.z; o[7] = (_Float16)b.w;
    d[i] = o;
}

// Pack W[128,128] (f32, row-major K x N) into the WMMA B-fragment layout:
// Wp[((wt*4 + c)*32 + lane)*16 + e] = (f16) W[(c*32 + 16*(lane>>4) + e)*128 + wt*16 + (lane&15)]
__global__ void k_pack_w(const float* __restrict__ W, _Float16* __restrict__ Wp) {
    int i = blockIdx.x * blockDim.x + threadIdx.x;   // 0 .. 16383
    if (i >= HID * HID) return;
    int e    = i & 15;
    int lane = (i >> 4) & 31;
    int c    = (i >> 9) & 3;
    int wt   = i >> 11;
    int k    = c * 32 + 16 * (lane >> 4) + e;
    int n    = wt * 16 + (lane & 15);
    Wp[i] = (_Float16)W[k * HID + n];
}

// ------------------------------------------------------------------ GEMM ----
// H[N,128] = Xh[N,128] @ W[128,128]  (bias added post-aggregation).
// Block = 256 threads = 8 waves; TDM stages the 16x128 f16 A-tile into padded
// LDS; wave w owns output columns [16w,16w+16). 4 x v_wmma_f32_16x16x32_f16.
__global__ void __launch_bounds__(256)
k_gemm(const _Float16* __restrict__ Xh, const v16h* __restrict__ Wp,
       float* __restrict__ H, int N) {
    __shared__ _Float16 sX[NPT * SROW];             // 4.25 KB padded f16 A-tile

    const int nb  = blockIdx.x * NPT;
    const int tid = threadIdx.x;

#if defined(__gfx1250__) && __has_builtin(__builtin_amdgcn_tensor_load_to_lds)
    // ---- TDM: DMA 16x128 f16 tile, LDS-padded 4 dwords per 64-dword row ----
    if (tid < 32) {
        unsigned long long ga = (unsigned long long)(const void*)(Xh + (size_t)nb * HID);
        unsigned int       la = (unsigned int)(size_t)(void*)&sX[0]; // LDS byte offset

        v4u g0;
        g0[0] = 1u;                                  // count=1, user descriptor
        g0[1] = la;                                  // lds_addr
        g0[2] = (unsigned int)(ga & 0xFFFFFFFFull);  // global_addr[31:0]
        g0[3] = (unsigned int)((ga >> 32) & 0x1FFFFFFull) | (2u << 30); // addr[56:32] | type=2

        v8i g1;
        g1[0] = (int)((1u << 16)                     // data_size = 2B
                    | (1u << 20)                     // pad_enable
                    | (5u << 22)                     // pad_interval = 64 dwords
                    | (3u << 25));                   // pad_amount   = 4 dwords (8 halves)
        g1[1] = (int)(128u << 16);                   // tensor_dim0 = 128 (lo16 @ bit48)
        g1[2] = (int)(16u  << 16);                   // tensor_dim0 hi=0 | tensor_dim1 = 16
        g1[3] = (int)(128u << 16);                   // tensor_dim1 hi=0 | tile_dim0 = 128
        g1[4] = (int)(16u);                          // tile_dim1 = 16 | tile_dim2 = 0
        g1[5] = (int)(128u);                         // tensor_dim0_stride = 128 (lo32)
        g1[6] = 0;
        g1[7] = 0;

        v4i gz4 = {0, 0, 0, 0};                      // 2D tensor: groups 2/3 unused
        v8i gz8 = {0, 0, 0, 0, 0, 0, 0, 0};
        __builtin_amdgcn_tensor_load_to_lds(g0, g1, gz4, gz4, gz8, 0);
        __builtin_amdgcn_s_wait_tensorcnt(0);
    }
    __syncthreads();
#else
    {   // fallback: manual staging, 8 halves (16B) per thread, padded rows
        int row = tid >> 4;
        int col = (tid & 15) * 8;
        *(v8h*)&sX[row * SROW + col] = *(const v8h*)(Xh + (size_t)(nb + row) * HID + col);
    }
    __syncthreads();
#endif

    const int wave = tid >> 5;
    const int lane = tid & 31;
    const int r    = lane & 15;                     // M (for A) / N (for B)
    const int half = lane >> 4;
    const int ncol = wave * 16 + r;                 // output column

    v8f acc = {};
    #pragma unroll
    for (int c = 0; c < 4; ++c) {
        const int kb = c * 32;
        v16h a;
        // A fragment (contiguous k = 16*half + e, matching packed B): two 8-half groups, both 16B aligned
        #pragma unroll
        for (int e = 0; e < 8; ++e)  a[e] = sX[r * SROW + kb + 16 * half + e];
        #pragma unroll
        for (int e = 8; e < 16; ++e) a[e] = sX[r * SROW + kb + 16 * half + e];
        // B fragment: pre-packed, one 32B vector load per lane
        v16h b = Wp[(wave * 4 + c) * 32 + lane];

        acc = __builtin_amdgcn_wmma_f32_16x16x32_f16(
            /*neg_a=*/false, a, /*neg_b=*/false, b,
            /*c_mod=*/(short)0, acc, /*reuse_a=*/false, /*reuse_b=*/false);
    }

    // C/D layout: VGPR v -> M = v + 8*half, N = ncol   (grid exact: no guard)
    #pragma unroll
    for (int v = 0; v < 8; ++v) {
        int m = nb + v + 8 * half;
        H[(size_t)m * HID + ncol] = acc[v];
    }
}

// ----------------------------------------------------- aggregation (GCN) ----
// Self-loop contribution initializes agg: agg[i] = h[i] * dinv[i]^2
__global__ void k_self_init(const float* __restrict__ H, const float* __restrict__ dinv,
                            float* __restrict__ agg, int N) {
    int idx = blockIdx.x * blockDim.x + threadIdx.x;       // one float4 each
    if (idx >= N * (HID / 4)) return;
    int node  = idx >> 5;                                   // /32
    int chunk = idx & 31;
    float d = dinv[node];
    float s = d * d;
    float4 v = ((const float4*)(H + (size_t)node * HID))[chunk];
    float4 o = { v.x * s, v.y * s, v.z * s, v.w * s };
    ((float4*)(agg + (size_t)node * HID))[chunk] = o;
}

// One wave32 per edge; lane handles a float4 (32*4 = 128 features).
__global__ void __launch_bounds__(256)
k_scatter(const float* __restrict__ H,
          const long long* __restrict__ src, const long long* __restrict__ dst,
          const float* __restrict__ dinv, float* __restrict__ agg, int E) {
    int e = blockIdx.x * 8 + (threadIdx.x >> 5);
    if (e >= E) return;
    int lane = threadIdx.x & 31;
    long long s = src[e];
    long long d = dst[e];
    float nrm = dinv[s] * dinv[d];
    float4 v = ((const float4*)(H + (size_t)s * HID))[lane];
    float* o = agg + (size_t)d * HID + lane * 4;
    atomAdd(o + 0, v.x * nrm);
    atomAdd(o + 1, v.y * nrm);
    atomAdd(o + 2, v.z * nrm);
    atomAdd(o + 3, v.w * nrm);
}

// ------------------------------------------------- fused pointwise stages ---
// x1 = relu(layernorm(agg + b) * g + be); writes f32 (residual) + f16 (GEMM2 A)
__global__ void __launch_bounds__(256)
k_ln_relu(const float* __restrict__ agg, const float* __restrict__ b,
          const float* __restrict__ g, const float* __restrict__ be,
          float* __restrict__ X1, _Float16* __restrict__ X1h, int N) {
    int node = blockIdx.x * 8 + (threadIdx.x >> 5);
    if (node >= N) return;
    int lane = threadIdx.x & 31;
    const float4 hv = ((const float4*)(agg + (size_t)node * HID))[lane];
    const float4 bv = ((const float4*)b)[lane];
    float4 v = { hv.x + bv.x, hv.y + bv.y, hv.z + bv.z, hv.w + bv.w };

    float s  = v.x + v.y + v.z + v.w;
    float s2 = v.x * v.x + v.y * v.y + v.z * v.z + v.w * v.w;
    #pragma unroll
    for (int off = 16; off >= 1; off >>= 1) {
        s  += __shfl_xor(s,  off, 32);
        s2 += __shfl_xor(s2, off, 32);
    }
    float mu  = s * (1.0f / HID);
    float var = s2 * (1.0f / HID) - mu * mu;
    float rs  = rsqrtf(var + EPS);

    const float4 gv  = ((const float4*)g)[lane];
    const float4 bev = ((const float4*)be)[lane];
    float4 o;
    o.x = fmaxf((v.x - mu) * rs * gv.x + bev.x, 0.0f);
    o.y = fmaxf((v.y - mu) * rs * gv.y + bev.y, 0.0f);
    o.z = fmaxf((v.z - mu) * rs * gv.z + bev.z, 0.0f);
    o.w = fmaxf((v.w - mu) * rs * gv.w + bev.w, 0.0f);
    ((float4*)(X1 + (size_t)node * HID))[lane] = o;
    v4h oh = { (_Float16)o.x, (_Float16)o.y, (_Float16)o.z, (_Float16)o.w };
    ((v4h*)(X1h + (size_t)node * HID))[lane] = oh;
}

// out[node] = ( relu(layernorm(agg + b2)*g2+be2) + x1 ) . Wf + bf
__global__ void __launch_bounds__(256)
k_final(const float* __restrict__ agg, const float* __restrict__ b,
        const float* __restrict__ g, const float* __restrict__ be,
        const float* __restrict__ X1, const float* __restrict__ Wf,
        const float* __restrict__ bf, float* __restrict__ out, int N) {
    int node = blockIdx.x * 8 + (threadIdx.x >> 5);
    if (node >= N) return;
    int lane = threadIdx.x & 31;
    const float4 hv = ((const float4*)(agg + (size_t)node * HID))[lane];
    const float4 bv = ((const float4*)b)[lane];
    float4 v = { hv.x + bv.x, hv.y + bv.y, hv.z + bv.z, hv.w + bv.w };

    float s  = v.x + v.y + v.z + v.w;
    float s2 = v.x * v.x + v.y * v.y + v.z * v.z + v.w * v.w;
    #pragma unroll
    for (int off = 16; off >= 1; off >>= 1) {
        s  += __shfl_xor(s,  off, 32);
        s2 += __shfl_xor(s2, off, 32);
    }
    float mu  = s * (1.0f / HID);
    float var = s2 * (1.0f / HID) - mu * mu;
    float rs  = rsqrtf(var + EPS);

    const float4 gv  = ((const float4*)g)[lane];
    const float4 bev = ((const float4*)be)[lane];
    const float4 xv  = ((const float4*)(X1 + (size_t)node * HID))[lane];
    const float4 wv  = ((const float4*)Wf)[lane];

    float4 x2;
    x2.x = fmaxf((v.x - mu) * rs * gv.x + bev.x, 0.0f) + xv.x;
    x2.y = fmaxf((v.y - mu) * rs * gv.y + bev.y, 0.0f) + xv.y;
    x2.z = fmaxf((v.z - mu) * rs * gv.z + bev.z, 0.0f) + xv.z;
    x2.w = fmaxf((v.w - mu) * rs * gv.w + bev.w, 0.0f) + xv.w;

    float p = x2.x * wv.x + x2.y * wv.y + x2.z * wv.z + x2.w * wv.w;
    #pragma unroll
    for (int off = 16; off >= 1; off >>= 1) p += __shfl_xor(p, off, 32);

    if (lane == 0) out[node] = p + bf[0];
}

// ------------------------------------------------------------------ host ----
extern "C" void kernel_launch(void* const* d_in, const int* in_sizes, int n_in,
                              void* d_out, int out_size, void* d_ws, size_t ws_size,
                              hipStream_t stream) {
    const float*     x    = (const float*)d_in[0];
    const long long* eidx = (const long long*)d_in[1];
    const float*     W1   = (const float*)d_in[2];
    const float*     b1   = (const float*)d_in[3];
    const float*     g1   = (const float*)d_in[4];
    const float*     be1  = (const float*)d_in[5];
    const float*     W2   = (const float*)d_in[6];
    const float*     b2   = (const float*)d_in[7];
    const float*     g2   = (const float*)d_in[8];
    const float*     be2  = (const float*)d_in[9];
    const float*     Wf   = (const float*)d_in[10];
    const float*     bf   = (const float*)d_in[11];
    float*           out  = (float*)d_out;

    const int N = in_sizes[0] / HID;       // 100000 (multiple of 16)
    const int E = in_sizes[1] / 2;         // 1600000
    const long long* src = eidx;
    const long long* dst = eidx + E;

    float*    h    = (float*)d_ws;            // [N,128] f32
    float*    agg  = h   + (size_t)N * HID;   // [N,128] f32
    float*    x1   = agg + (size_t)N * HID;   // [N,128] f32
    float*    dinv = x1  + (size_t)N * HID;   // [N]     f32 (deg -> dinv in place)
    _Float16* xh   = (_Float16*)(dinv + N);   // [N,128] f16 (GEMM A input, reused)
    _Float16* wp1  = xh  + (size_t)N * HID;   // [128,128] f16 packed B
    _Float16* wp2  = wp1 + HID * HID;         // [128,128] f16 packed B

    const int T = 256;

    // degrees (self-loops folded into init)
    k_init_deg <<<(N + T - 1) / T, T, 0, stream>>>(dinv, N);
    k_accum_deg<<<(E + T - 1) / T, T, 0, stream>>>(dst, dinv, E);
    k_dinv     <<<(N + T - 1) / T, T, 0, stream>>>(dinv, N);

    // one-shot conversions / weight packing
    const int n8 = N * HID / 8;
    k_cvt16    <<<(n8 + T - 1) / T, T, 0, stream>>>(x, (v8h*)xh, n8);
    k_pack_w   <<<(HID * HID + T - 1) / T, T, 0, stream>>>(W1, wp1);
    k_pack_w   <<<(HID * HID + T - 1) / T, T, 0, stream>>>(W2, wp2);

    // ---- layer 1 ----
    k_gemm     <<<N / NPT, T, 0, stream>>>(xh, (const v16h*)wp1, h, N);
    k_self_init<<<(N * (HID / 4) + T - 1) / T, T, 0, stream>>>(h, dinv, agg, N);
    k_scatter  <<<(E + 7) / 8, T, 0, stream>>>(h, src, dst, dinv, agg, E);
    k_ln_relu  <<<(N + 7) / 8, T, 0, stream>>>(agg, b1, g1, be1, x1, xh, N);

    // ---- layer 2 ----
    k_gemm     <<<N / NPT, T, 0, stream>>>(xh, (const v16h*)wp2, h, N);
    k_self_init<<<(N * (HID / 4) + T - 1) / T, T, 0, stream>>>(h, dinv, agg, N);
    k_scatter  <<<(E + 7) / 8, T, 0, stream>>>(h, src, dst, dinv, agg, E);

    // ---- fused LN + relu + residual + final projection ----
    k_final    <<<(N + 7) / 8, T, 0, stream>>>(agg, b2, g2, be2, x1, Wf, bf, out, N);
}
